// HMMSmoother_17514876633778
// MI455X (gfx1250) — matrix operand order repaired and checked by reading, not compiled, not run on hardware
//
#include <hip/hip_runtime.h>
#include <stdint.h>

// HMM forward-backward smoother for MI455X (gfx1250, wave32, WMMA + TDM).
// B=16 batches map onto the WMMA M dimension; C=170 padded to N=176 (11 tiles,
// one wave each -> 352 threads, fully uniform WMMA) and K=192 (6 slices of 32).
// Transition matrix resident in LDS as f16; B fragments preloaded into VGPRs
// once and reused for all 8192 steps. Per-step logits tiles are streamed into
// a double-buffered LDS staging area by the Tensor Data Mover: issue for step
// t+1 at the top of step t, wait at the bottom -> DMA overlaps the whole step.
// Row reductions are intra-wave shuffle butterflies (no LDS, no barriers):
// 3 workgroup barriers per scan step total.

#define C_   170
#define CP_  176   // padded N
#define KP_  192   // padded K
#define B_   16
#define T_   8192
#define NSL  6     // K slices of 32
#define NTH  352   // 11 waves (wave32): one N-tile per wave

typedef _Float16 v16h __attribute__((ext_vector_type(16)));
typedef _Float16 v8h  __attribute__((ext_vector_type(8)));
typedef float    v8f  __attribute__((ext_vector_type(8)));
typedef unsigned int v4u __attribute__((ext_vector_type(4)));
typedef int          v8i __attribute__((ext_vector_type(8)));
typedef int          v4i __attribute__((ext_vector_type(4)));

// ---- Tensor Data Mover: load a 16 x 170 f32 tile (rows strided by T*C) into LDS.
// D# per CDNA5 ISA ch.8: group0 = {count/type/lds_addr/global_addr},
// group1 = {data_size, tensor dims/strides, tile dims}, groups 2/3 unused (2D).
__device__ __forceinline__ void tdm_load_logits_tile(const float* gptr, unsigned lds_off) {
  const unsigned long long ga = (unsigned long long)(uintptr_t)gptr;
  const v4u g0 = {
    1u,                                                   // count=1, user descriptor
    lds_off,                                              // lds_addr (bytes)
    (unsigned)ga,                                         // global_addr[31:0]
    (unsigned)((ga >> 32) & 0x1FFFFFFu) | (2u << 30)      // global_addr[56:32] | type=2
  };
  const v8i g1 = {
    (int)(2u << 16),             // workgroup_mask=0, data_size=2 (4 bytes)
    (int)((unsigned)C_ << 16),   // tensor_dim0 = 170 (low16 at bits 63:48)
    (int)((unsigned)B_ << 16),   // tensor_dim1 = 16  (low16 at bits 95:80)
    (int)((unsigned)C_ << 16),   // tile_dim0  = 170  (bits 127:112)
    B_,                          // tile_dim1 = 16, tile_dim2 = 0
    (int)(T_ * C_),              // tensor_dim0_stride low32 (elements)
    0,                           // stride0 hi16 = 0, stride1 lo16 = 0
    0                            // stride1 hi32 = 0
  };
  const v4i gz4 = { 0, 0, 0, 0 };
  const v8i gz8 = { 0, 0, 0, 0, 0, 0, 0, 0 };
  __builtin_amdgcn_tensor_load_to_lds(g0, g1, gz4, gz4, gz8, 0);
}

// B operand (K x N tile), M stored column-major: M[n*KP_ + k].
__device__ __forceinline__ v16h load_b_frag(const _Float16* M, int tile, int kb, int lane) {
  const int col = tile * 16 + (lane & 15);
  const int kr  = kb + ((lane >> 4) << 4);
  return *(const v16h*)(M + col * KP_ + kr);
}

// A operand (16 x 32 slice), A stored row-major: A[m*KP_ + k].
__device__ __forceinline__ v16h load_a_frag(const _Float16* A, int kb, int lane) {
  const int m  = lane & 15;
  const int k0 = kb + ((lane >> 4) << 4);
  const v8h* p = (const v8h*)(A + m * KP_ + k0);
  const v8h lo = p[0];  // K = k0..k0+7
  const v8h hi = p[1];  // K = k0+8..k0+15
  return __builtin_shufflevector(lo, hi, 0,1,2,3,4,5,6,7,8,9,10,11,12,13,14,15);
}

// C/D layout: VGPR r holds row (r + 8*(lane>=16)), col = tile*16 + (lane&15).
__device__ __forceinline__ void store_tile(float* Af32, const float* P, v8f acc,
                                           int tile, int lane, bool mulP) {
  const int n  = tile * 16 + (lane & 15);
  const int b0 = (lane >> 4) << 3;
#pragma unroll
  for (int r = 0; r < 8; ++r) {
    float v = acc[r];
    if (mulP) v *= P[(b0 + r) * CP_ + n];
    Af32[(b0 + r) * CP_ + n] = v;
  }
}

// All 12 ds_load_b128 issued up front, then 6 back-to-back WMMAs on one accumulator.
__device__ __forceinline__ v8f gemm_state(const _Float16* Af16, const v16h bf[NSL], int lane) {
  v16h af[NSL];
#pragma unroll
  for (int s = 0; s < NSL; ++s) af[s] = load_a_frag(Af16, 32 * s, lane);
  v8f acc = {};
#pragma unroll
  for (int s = 0; s < NSL; ++s)
    acc = __builtin_amdgcn_wmma_f32_16x16x32_f16(false, af[s], false, bf[s],
                                                 (short)0, acc, false, false);
  return acc;
}

// butterfly reduce over the 16 lanes servicing one batch row (intra-wave, no barriers)
__device__ __forceinline__ float row_max16(float m) {
#pragma unroll
  for (int o = 8; o >= 1; o >>= 1) m = fmaxf(m, __shfl_xor(m, o, 16));
  return m;
}
__device__ __forceinline__ float row_sum16(float s) {
#pragma unroll
  for (int o = 8; o >= 1; o >>= 1) s += __shfl_xor(s, o, 16);
  return s;
}

__global__ __launch_bounds__(NTH)
void hmm_fb_pass_kernel(const float* __restrict__ logits,
                        const float* __restrict__ trans,
                        const float* __restrict__ initd,
                        float* __restrict__ outp)
{
  __shared__ _Float16 Mlds[CP_ * KP_];   // transition matrix (transposed per dir), f16
  __shared__ _Float16 Af16[B_ * KP_];    // GEMM A input (state), f16, zero-padded
  __shared__ float    Af32[B_ * CP_];    // state in f32
  __shared__ float    Plds[B_ * CP_];    // emission p, f32, zero-padded
  __shared__ float    Llds[2][B_ * C_];  // TDM double buffer: logits tiles

  const int tid  = threadIdx.x;
  const int lane = tid & 31;
  const int wid  = tid >> 5;
  const int dir  = blockIdx.x;  // 0 = forward (alpha), 1 = backward (beta)

  // wave-uniform scalar wave id -> scalar branches (TDM ignores EXEC, so the
  // issuing wave must be selected by an s_cbranch, not an exec mask).
  const int swid = __builtin_amdgcn_readfirstlane(wid);

  // ---- one-time LDS init ----
  for (int i = tid; i < CP_ * KP_; i += NTH) {
    const int n = i / KP_, k = i % KP_;
    float v = 0.f;
    // forward: Anew = A @ T      -> B(k,n) = T[k][n]
    // backward: Bnew = W @ T^T   -> B(k,n) = T[n][k]
    if (n < C_ && k < C_) v = dir ? trans[n * C_ + k] : trans[k * C_ + n];
    Mlds[i] = (_Float16)v;
  }
  for (int i = tid; i < B_ * KP_; i += NTH) Af16[i] = (_Float16)0.f;
  for (int i = tid; i < B_ * CP_; i += NTH) { Plds[i] = 0.f; Af32[i] = dir ? 1.f : 0.f; }

  // prologue TDM: stage the first logits tile and wait for it before the
  // barrier that opens the main loop.
  if (swid == 0) {
    const int r0 = dir ? (T_ - 1) : 0;
    tdm_load_logits_tile(logits + (size_t)r0 * C_,
                         (unsigned)(uintptr_t)(&Llds[r0 & 1][0]));
    __builtin_amdgcn_s_wait_tensorcnt(0);
  }
  __syncthreads();

  // ---- preload B fragments into VGPRs (reused for all 8192 steps) ----
  const int tile = wid;  // 0..10
  v16h bf[NSL];
#pragma unroll
  for (int s = 0; s < NSL; ++s) bf[s] = load_b_frag(Mlds, tile, 32 * s, lane);

  const int b = tid >> 4;   // batch row this thread services in scalar phases (tid<256)
  const int j = tid & 15;
  unsigned short* outh = (unsigned short*)outp;  // alpha -> low half, beta -> high half

  if (dir == 0) {
    // ====================== forward pass ======================
    for (int t = 0; t < T_; ++t) {
      // fire-and-forget: stage tile t+1; it has the whole step to land
      if (swid == 0 && t + 1 < T_)
        tdm_load_logits_tile(logits + (size_t)(t + 1) * C_,
                             (unsigned)(uintptr_t)(&Llds[(t + 1) & 1][0]));

      // --- phase P: p = exp(logit - rowmax) from the staged LDS tile ---
      if (tid < 256) {
        const float* lr = &Llds[t & 1][b * C_];
        float xv[11];
        float m = -1e30f;
#pragma unroll
        for (int i = 0; i < 11; ++i) {
          const int c = j + 16 * i;
          const float x = (c < C_) ? lr[c] : -1e30f;
          xv[i] = x;
          m = fmaxf(m, x);
        }
        const float rm = row_max16(m);
#pragma unroll
        for (int i = 0; i < 11; ++i) {
          const int c = j + 16 * i;
          if (c < C_) Plds[b * CP_ + c] = __expf(xv[i] - rm);
        }
      }
      __syncthreads();

      // --- phase G: A_new = (A_prev @ T) * p  (WMMA, uniform over all 11 waves) ---
      if (t == 0) {
        if (tid < 256) {
#pragma unroll
          for (int i = 0; i < 11; ++i) {
            const int c = j + 16 * i;
            if (c < C_) Af32[b * CP_ + c] = initd[c] * Plds[b * CP_ + c];
          }
        }
      } else {
        const v8f acc = gemm_state(Af16, bf, lane);
        store_tile(Af32, Plds, acc, tile, lane, true);
      }
      __syncthreads();

      // --- phase N: normalize, refresh f16 state, emit alpha (low halfword) ---
      if (tid < 256) {
        float s = 0.f;
#pragma unroll
        for (int i = 0; i < 11; ++i) { const int c = j + 16 * i; if (c < C_) s += Af32[b * CP_ + c]; }
        const float tot = row_sum16(s);
        const float inv = (tot > 0.f) ? (1.f / tot) : 0.f;
        const size_t base = ((size_t)b * T_ + t) * C_;
#pragma unroll
        for (int i = 0; i < 11; ++i) {
          const int c = j + 16 * i;
          if (c < C_) {
            const float v = Af32[b * CP_ + c] * inv;
            const _Float16 h = (_Float16)v;
            Af16[b * KP_ + c] = h;
            outh[2 * (base + c)] = __builtin_bit_cast(unsigned short, h);
          }
        }
      }
      // ensure tile t+1 landed before the barrier that opens step t+1
      if (swid == 0) __builtin_amdgcn_s_wait_tensorcnt(0);
      __syncthreads();
    }
  } else {
    // ====================== backward pass ======================
    if (tid < 256) {   // beta_{T-1} = 1
      const size_t base = ((size_t)b * T_ + (T_ - 1)) * C_;
#pragma unroll
      for (int i = 0; i < 11; ++i) {
        const int c = j + 16 * i;
        if (c < C_) outh[2 * (base + c) + 1] = (unsigned short)0x3C00u;  // f16 1.0
      }
    }
    for (int tt = T_ - 2; tt >= 0; --tt) {
      // fire-and-forget: stage row tt (needed next iteration)
      if (swid == 0 && tt >= 1)
        tdm_load_logits_tile(logits + (size_t)tt * C_,
                             (unsigned)(uintptr_t)(&Llds[tt & 1][0]));

      // --- phase P: W = p_{tt+1} * beta_{tt+1}  (into f16 A buffer) ---
      if (tid < 256) {
        const float* lr = &Llds[(tt + 1) & 1][b * C_];
        float xv[11];
        float m = -1e30f;
#pragma unroll
        for (int i = 0; i < 11; ++i) {
          const int c = j + 16 * i;
          const float x = (c < C_) ? lr[c] : -1e30f;
          xv[i] = x;
          m = fmaxf(m, x);
        }
        const float rm = row_max16(m);
#pragma unroll
        for (int i = 0; i < 11; ++i) {
          const int c = j + 16 * i;
          if (c < C_)
            Af16[b * KP_ + c] = (_Float16)(__expf(xv[i] - rm) * Af32[b * CP_ + c]);
        }
      }
      __syncthreads();

      // --- phase G: beta_tt = W @ T^T  (WMMA, uniform over all 11 waves) ---
      {
        const v8f acc = gemm_state(Af16, bf, lane);
        store_tile(Af32, Plds, acc, tile, lane, false);
      }
      __syncthreads();

      // --- phase N: normalize beta, emit (high halfword) ---
      if (tid < 256) {
        float s = 0.f;
#pragma unroll
        for (int i = 0; i < 11; ++i) { const int c = j + 16 * i; if (c < C_) s += Af32[b * CP_ + c]; }
        const float tot = row_sum16(s);
        const float inv = (tot > 0.f) ? (1.f / tot) : 0.f;
        const size_t base = ((size_t)b * T_ + tt) * C_;
#pragma unroll
        for (int i = 0; i < 11; ++i) {
          const int c = j + 16 * i;
          if (c < C_) {
            const float v = Af32[b * CP_ + c] * inv;
            Af32[b * CP_ + c] = v;  // next step reads normalized beta
            outh[2 * (base + c) + 1] =
                __builtin_bit_cast(unsigned short, (_Float16)v);
          }
        }
      }
      if (swid == 0) __builtin_amdgcn_s_wait_tensorcnt(0);
      __syncthreads();
    }
  }
}

// gamma = normalize(alpha * beta) per (b,t) row; unpack the two f16 halves,
// overwrite with the f32 posterior.
__global__ __launch_bounds__(256)
void hmm_combine_kernel(float* outp)
{
  __shared__ float sdata[256];
  const int tid = threadIdx.x;
  const size_t base = (size_t)blockIdx.x * C_;

  float val = 0.f;
  if (tid < C_) {
    const unsigned int w = ((const unsigned int*)outp)[base + tid];
    const _Float16 lo = __builtin_bit_cast(_Float16, (unsigned short)(w & 0xFFFFu));
    const _Float16 hi = __builtin_bit_cast(_Float16, (unsigned short)(w >> 16));
    val = (float)lo * (float)hi;
  }
  sdata[tid] = val;
  __syncthreads();
#pragma unroll
  for (int off = 128; off > 0; off >>= 1) {
    if (tid < off) sdata[tid] += sdata[tid + off];
    __syncthreads();
  }
  const float tot = sdata[0];
  const float inv = (tot > 0.f) ? (1.f / tot) : 0.f;
  if (tid < C_) outp[base + tid] = val * inv;
}

extern "C" void kernel_launch(void* const* d_in, const int* in_sizes, int n_in,
                              void* d_out, int out_size, void* d_ws, size_t ws_size,
                              hipStream_t stream)
{
  (void)in_sizes; (void)n_in; (void)out_size; (void)d_ws; (void)ws_size;
  const float* logits = (const float*)d_in[0];
  const float* trans  = (const float*)d_in[1];
  const float* initd  = (const float*)d_in[2];
  float* outp = (float*)d_out;

  // block 0: forward chain, block 1: backward chain (independent, concurrent)
  hmm_fb_pass_kernel<<<dim3(2), dim3(NTH), 0, stream>>>(logits, trans, initd, outp);
  // elementwise combine + per-row softmax-style normalization
  hmm_combine_kernel<<<dim3(B_ * T_), dim3(256), 0, stream>>>(outp);
}
